// LocationSensitiveAttention_28338194219452
// MI455X (gfx1250) — matrix operand order repaired and checked by reading, not compiled, not run on hardware
//
#include <hip/hip_runtime.h>
#include <hip/hip_bf16.h>
#include <math.h>

typedef __attribute__((ext_vector_type(16))) _Float16 v16h;
typedef __attribute__((ext_vector_type(8)))  _Float16 v8h;
typedef __attribute__((ext_vector_type(8)))  float    v8f;

constexpr int B_   = 64;
constexpr int T_   = 2048;
constexpr int DIN_ = 1024;
constexpr int E_   = 512;
constexpr int A_   = 128;
constexpr int NF_  = 32;
constexpr int K_   = 31;
constexpr int PAD_ = 15;

// ---------------------------------------------------------------------------
// Kernel 0: one-time f32 -> f16 conversion of Wv [A,E] and Wl [A,NF]
// ---------------------------------------------------------------------------
__global__ void lsa_cvt_weights(const float* __restrict__ Wv,
                                const float* __restrict__ Wl,
                                _Float16* __restrict__ Wv_h,
                                _Float16* __restrict__ Wl_h) {
    const int i = blockIdx.x * blockDim.x + threadIdx.x;
    if (i < A_ * E_) Wv_h[i] = (_Float16)Wv[i];
    if (i < A_ * NF_) Wl_h[i] = (_Float16)Wl[i];
}

// ---------------------------------------------------------------------------
// Kernel 1: q_proj[b,a] = dot(query[b,:], Wq[a,:])   (B x A = 64 x 128)
// ---------------------------------------------------------------------------
__global__ void lsa_qproj(const float* __restrict__ q,
                          const float* __restrict__ Wq,
                          float* __restrict__ qp) {
    __shared__ float qs[DIN_];
    const int b = blockIdx.x;
    for (int i = threadIdx.x; i < DIN_; i += blockDim.x) qs[i] = q[b * DIN_ + i];
    __syncthreads();
    const int a = threadIdx.x;              // blockDim.x == A_
    const float* wr = Wq + (size_t)a * DIN_;
    float acc = 0.f;
#pragma unroll 8
    for (int i = 0; i < DIN_; ++i) acc = fmaf(qs[i], wr[i], acc);
    qp[b * A_ + a] = acc;
}

// ---------------------------------------------------------------------------
// Kernel 2: conv_out[b,t,f] (stored f16, [B][T][NF]) =
//   sum_{c,k} conv_w[f,c,k] * loc_in[b,c,t+k-PAD]  (zero padded)
// ---------------------------------------------------------------------------
__global__ void lsa_conv(const float* __restrict__ aw,
                         const float* __restrict__ ac,
                         const float* __restrict__ cw,
                         _Float16* __restrict__ out) {
    __shared__ float w[NF_ * 2 * K_];       // 1984 floats
    for (int i = threadIdx.x; i < NF_ * 2 * K_; i += blockDim.x) w[i] = cw[i];
    __syncthreads();

    const int b = blockIdx.x;
    const int t = blockIdx.y * blockDim.x + threadIdx.x;

    float wa[K_], wc[K_];
#pragma unroll
    for (int k = 0; k < K_; ++k) {
        const int tt = t + k - PAD_;
        const bool in = (tt >= 0) && (tt < T_);
        wa[k] = in ? aw[b * T_ + tt] : 0.f;
        wc[k] = in ? ac[b * T_ + tt] : 0.f;
    }
    _Float16* op = out + ((size_t)b * T_ + t) * NF_;
#pragma unroll 4
    for (int f = 0; f < NF_; ++f) {
        const float* wf = w + f * 2 * K_;
        float acc = 0.f;
#pragma unroll
        for (int k = 0; k < K_; ++k) {
            acc = fmaf(wf[k],      wa[k], acc);
            acc = fmaf(wf[K_ + k], wc[k], acc);
        }
        op[f] = (_Float16)acc;
    }
}

// ---------------------------------------------------------------------------
// Kernel 3: WMMA scores.  One wave computes a 32(t) x 128(a) tile:
//   pre[t,a] = sum_e values[b,t,e]*Wv[a,e] + sum_f conv[b,t,f]*Wl[a,f] + qp[b,a]
//   scores[b,t] = sum_a tanh(pre[t,a]) * Ws[a]
// Block = 256 threads = 8 waves, each wave owns two 16-row t-tiles (32 rows),
// doubling B-operand reuse.  B operands are pre-converted f16 (no cvt in loop).
// ---------------------------------------------------------------------------
__global__ void __launch_bounds__(256)
lsa_scores(const float* __restrict__ values,
           const _Float16* __restrict__ Wv_h,
           const _Float16* __restrict__ Wl_h,
           const _Float16* __restrict__ conv_h,
           const float* __restrict__ qp,
           const float* __restrict__ Ws,
           float* __restrict__ scores) {
    const int lane = threadIdx.x & 31;
    const int wave = threadIdx.x >> 5;
    const int b    = blockIdx.x;
    const int t0   = (blockIdx.y * 8 + wave) * 32;   // 32 t-rows per wave

    const int mr = lane & 15;   // A-operand: row (t); C-operand: column (a)
    const int hs = lane >> 4;   // half-select per 16-bit WMMA lane layout

    const float* arow0 = values + ((size_t)b * T_ + (t0 + mr)) * E_;
    const float* arow1 = arow0 + (size_t)16 * E_;

    v8f c[2][8] = {};           // [m-tile][n-tile] 16x16 f32 accumulators

    // ---- main K loop over E=512 (16 steps of K=32, f16) ----
    for (int ks = 0; ks < E_ / 32; ++ks) {
        const int kb = ks * 32;
        // A: 16x32 f16. lane<16: K=0..15; lane>=16: K=16..31
        v16h a0, a1;
        {
            const float4 x0 = *(const float4*)(arow0 + kb + hs * 16);
            const float4 x1 = *(const float4*)(arow0 + kb + hs * 16 + 4);
            const float4 y0 = *(const float4*)(arow0 + kb + hs * 16 + 8);
            const float4 y1 = *(const float4*)(arow0 + kb + hs * 16 + 12);
            a0[0]=(_Float16)x0.x; a0[1]=(_Float16)x0.y; a0[2]=(_Float16)x0.z; a0[3]=(_Float16)x0.w;
            a0[4]=(_Float16)x1.x; a0[5]=(_Float16)x1.y; a0[6]=(_Float16)x1.z; a0[7]=(_Float16)x1.w;
            a0[8]=(_Float16)y0.x; a0[9]=(_Float16)y0.y; a0[10]=(_Float16)y0.z; a0[11]=(_Float16)y0.w;
            a0[12]=(_Float16)y1.x; a0[13]=(_Float16)y1.y; a0[14]=(_Float16)y1.z; a0[15]=(_Float16)y1.w;
        }
        {
            const float4 x0 = *(const float4*)(arow1 + kb + hs * 16);
            const float4 x1 = *(const float4*)(arow1 + kb + hs * 16 + 4);
            const float4 y0 = *(const float4*)(arow1 + kb + hs * 16 + 8);
            const float4 y1 = *(const float4*)(arow1 + kb + hs * 16 + 12);
            a1[0]=(_Float16)x0.x; a1[1]=(_Float16)x0.y; a1[2]=(_Float16)x0.z; a1[3]=(_Float16)x0.w;
            a1[4]=(_Float16)x1.x; a1[5]=(_Float16)x1.y; a1[6]=(_Float16)x1.z; a1[7]=(_Float16)x1.w;
            a1[8]=(_Float16)y0.x; a1[9]=(_Float16)y0.y; a1[10]=(_Float16)y0.z; a1[11]=(_Float16)y0.w;
            a1[12]=(_Float16)y1.x; a1[13]=(_Float16)y1.y; a1[14]=(_Float16)y1.z; a1[15]=(_Float16)y1.w;
        }

#pragma unroll
        for (int nt = 0; nt < 8; ++nt) {
            // B: 32x16 f16, pre-converted. lane<16: col=lane, K=kb+0..15;
            // lane>=16: K=kb+16..31.  32B aligned contiguous -> 2x b128.
            const int acol = nt * 16 + mr;
            const v16h bm = *(const v16h*)(Wv_h + (size_t)acol * E_ + kb + hs * 16);
            c[0][nt] = __builtin_amdgcn_wmma_f32_16x16x32_f16(
                false, a0, false, bm, (short)0, c[0][nt], false, false);
            c[1][nt] = __builtin_amdgcn_wmma_f32_16x16x32_f16(
                false, a1, false, bm, (short)0, c[1][nt], false, false);
        }
    }

    // ---- location term: one extra K=32 step (NF=32): conv_out @ Wl.T ----
    {
        const _Float16* cp0 = conv_h + ((size_t)b * T_ + (t0 + mr)) * NF_;
        const _Float16* cp1 = cp0 + (size_t)16 * NF_;
        v16h a0, a1;
        {
            const v8h u = *(const v8h*)(cp0 + hs * 16);
            const v8h v = *(const v8h*)(cp0 + hs * 16 + 8);
#pragma unroll
            for (int i = 0; i < 8; ++i) { a0[i] = u[i]; a0[8 + i] = v[i]; }
        }
        {
            const v8h u = *(const v8h*)(cp1 + hs * 16);
            const v8h v = *(const v8h*)(cp1 + hs * 16 + 8);
#pragma unroll
            for (int i = 0; i < 8; ++i) { a1[i] = u[i]; a1[8 + i] = v[i]; }
        }
#pragma unroll
        for (int nt = 0; nt < 8; ++nt) {
            const int acol = nt * 16 + mr;
            const v16h bm = *(const v16h*)(Wl_h + acol * NF_ + hs * 16);
            c[0][nt] = __builtin_amdgcn_wmma_f32_16x16x32_f16(
                false, a0, false, bm, (short)0, c[0][nt], false, false);
            c[1][nt] = __builtin_amdgcn_wmma_f32_16x16x32_f16(
                false, a1, false, bm, (short)0, c[1][nt], false, false);
        }
    }

    // ---- epilogue: +q_proj, tanh, *Ws, reduce over a ----
    // C layout: lane (0-15|16-31) holds N=lane&15; VGPR v holds M=v (+8 if lane>=16)
    float sacc[2][8];
#pragma unroll
    for (int mt = 0; mt < 2; ++mt)
#pragma unroll
        for (int v = 0; v < 8; ++v) sacc[mt][v] = 0.f;

#pragma unroll
    for (int nt = 0; nt < 8; ++nt) {
        const int acol = nt * 16 + mr;
        const float qv = qp[b * A_ + acol];
        const float wv = Ws[acol];
#pragma unroll
        for (int mt = 0; mt < 2; ++mt)
#pragma unroll
            for (int v = 0; v < 8; ++v) {
                const float pre = c[mt][nt][v] + qv;
                sacc[mt][v] = fmaf(tanhf(pre), wv, sacc[mt][v]);
            }
    }
#pragma unroll
    for (int mt = 0; mt < 2; ++mt)
#pragma unroll
        for (int v = 0; v < 8; ++v) {
            float s = sacc[mt][v];
            s += __shfl_xor(s, 1, 32);
            s += __shfl_xor(s, 2, 32);
            s += __shfl_xor(s, 4, 32);
            s += __shfl_xor(s, 8, 32);      // sum over the 16 N-lanes in each half
            if (mr == 0) scores[b * T_ + t0 + mt * 16 + v + 8 * hs] = s;
        }
}

// ---------------------------------------------------------------------------
// Kernel 4: per-batch softmax over T, then context[b,e] = sum_t attn*values
// ---------------------------------------------------------------------------
__global__ void __launch_bounds__(512)
lsa_softmax_ctx(const float* __restrict__ scores,
                const float* __restrict__ values,
                float* __restrict__ context,
                float* __restrict__ attn_out) {
    __shared__ float sm[T_];
    __shared__ float red[512];
    const int b = blockIdx.x, tid = threadIdx.x;

    float mx = -INFINITY;
    for (int t = tid; t < T_; t += 512) {
        const float s = scores[b * T_ + t];
        sm[t] = s;
        mx = fmaxf(mx, s);
    }
    red[tid] = mx;
    __syncthreads();
    for (int s = 256; s > 0; s >>= 1) {
        if (tid < s) red[tid] = fmaxf(red[tid], red[tid + s]);
        __syncthreads();
    }
    const float mxv = red[0];
    __syncthreads();

    float sum = 0.f;
    for (int t = tid; t < T_; t += 512) {
        const float e = __expf(sm[t] - mxv);
        sm[t] = e;
        sum += e;
    }
    red[tid] = sum;
    __syncthreads();
    for (int s = 256; s > 0; s >>= 1) {
        if (tid < s) red[tid] += red[tid + s];
        __syncthreads();
    }
    const float inv = 1.f / red[0];
    __syncthreads();

    for (int t = tid; t < T_; t += 512) {
        const float a = sm[t] * inv;
        sm[t] = a;
        attn_out[b * T_ + t] = a;
    }
    __syncthreads();

    // context: one e-column per thread (E_ == 512 == blockDim)
    float acc = 0.f;
    const float* vp = values + (size_t)b * T_ * E_ + tid;
#pragma unroll 4
    for (int t = 0; t < T_; ++t) acc = fmaf(sm[t], vp[(size_t)t * E_], acc);
    context[b * E_ + tid] = acc;
}

// ---------------------------------------------------------------------------
extern "C" void kernel_launch(void* const* d_in, const int* in_sizes, int n_in,
                              void* d_out, int out_size, void* d_ws, size_t ws_size,
                              hipStream_t stream) {
    const float* query  = (const float*)d_in[0];
    const float* values = (const float*)d_in[1];
    const float* attn_w = (const float*)d_in[2];
    const float* attn_c = (const float*)d_in[3];
    const float* Wq     = (const float*)d_in[4];
    const float* Wv     = (const float*)d_in[5];
    const float* Ws     = (const float*)d_in[6];
    const float* cw     = (const float*)d_in[7];
    const float* Wl     = (const float*)d_in[8];

    float* context  = (float*)d_out;              // [B, E]
    float* attn_out = (float*)d_out + B_ * E_;    // [B, T]

    float*    qp     = (float*)d_ws;                          // B*A floats
    float*    scores = qp + B_ * A_;                          // B*T floats
    _Float16* conv_h = (_Float16*)(scores + B_ * T_);         // B*T*NF halfs
    _Float16* Wv_h   = conv_h + (size_t)B_ * T_ * NF_;        // A*E halfs
    _Float16* Wl_h   = Wv_h + (size_t)A_ * E_;                // A*NF halfs

    lsa_cvt_weights<<<dim3((A_ * E_ + 255) / 256), dim3(256), 0, stream>>>(
        Wv, Wl, Wv_h, Wl_h);
    lsa_qproj<<<dim3(B_), dim3(A_), 0, stream>>>(query, Wq, qp);
    lsa_conv<<<dim3(B_, T_ / 256), dim3(256), 0, stream>>>(attn_w, attn_c, cw, conv_h);
    lsa_scores<<<dim3(B_, T_ / 256), dim3(256), 0, stream>>>(
        values, Wv_h, Wl_h, conv_h, qp, Ws, scores);
    lsa_softmax_ctx<<<dim3(B_), dim3(512), 0, stream>>>(scores, values, context, attn_out);
}
